// Transformer_16990890623294
// MI455X (gfx1250) — compile-verified, matrix operand order checked by reading, not run on hardware
//
#include <hip/hip_runtime.h>
#include <hip/hip_bf16.h>
#include <math.h>

typedef __attribute__((ext_vector_type(16))) _Float16 v16h;
typedef __attribute__((ext_vector_type(8)))  float    v8f;
typedef unsigned int uint32x4 __attribute__((ext_vector_type(4)));
typedef int          int32x4  __attribute__((ext_vector_type(4)));
typedef int          int32x8  __attribute__((ext_vector_type(8)));

#define BB   2
#define SSQ  512
#define VVC  32000
#define DDM  768
#define FFH  2048
#define LLN  4
#define NH   12
#define HDIM 64
#define MMR  (BB*SSQ)          // 1024 rows
#define YLD  (SSQ + 8 - 1)     // 519
#define FCSN 8

// Tensor Data Mover path: probe-confirmed builtins (ROCm7.2: 5-arg, clang-23: 6-arg)
#if defined(__HIP_DEVICE_COMPILE__) && \
    __has_builtin(__builtin_amdgcn_tensor_load_to_lds) && \
    __has_builtin(__builtin_amdgcn_s_wait_tensorcnt)
#define USE_TDM 1
#else
#define USE_TDM 0
#endif

#if USE_TDM
// Issue one TDM descriptor: copy a (rows x 32 f32) tile, row stride = stride_elems,
// from global memory into LDS at lds_dst (compact row-major, 128B per row).
__device__ __forceinline__ void tdm_load_2d(void* lds_dst, const void* gsrc,
                                            int rows, int stride_elems) {
  unsigned long long ga = (unsigned long long)gsrc;
  unsigned lds = (unsigned)(unsigned long long)lds_dst;  // generic->LDS: low 32 bits
  const unsigned TD = 0x40000000u;                       // huge tensor dims: no OOB clip
  uint32x4 g0;
  g0[0] = 1u;                                            // count=1 (valid user D#)
  g0[1] = lds;                                           // D#.lds_addr (bytes)
  g0[2] = (unsigned)(ga & 0xFFFFFFFFu);                  // global_addr[31:0]
  g0[3] = (unsigned)((ga >> 32) & 0x01FFFFFFu) | (2u << 30);  // addr[56:32] | type=2
  int32x8 g1;
  g1[0] = 0x00020000;                                    // wg_mask=0, data_size=4B
  g1[1] = (int)((TD & 0xFFFFu) << 16);                   // tensor_dim0[15:0]
  g1[2] = (int)(((TD >> 16) & 0xFFFFu) | ((TD & 0xFFFFu) << 16));  // d0 hi | d1 lo
  g1[3] = (int)(((TD >> 16) & 0xFFFFu) | (32u << 16));   // d1 hi | tile_dim0=32
  g1[4] = rows;                                          // tile_dim1 | tile_dim2=0
  g1[5] = stride_elems;                                  // tensor_dim0_stride lo
  g1[6] = 0;                                             // stride hi | dim1_stride lo
  g1[7] = 0;
  int32x4 z4 = {0, 0, 0, 0};
#if __clang_major__ >= 23
  int32x8 z8 = {0, 0, 0, 0, 0, 0, 0, 0};
  __builtin_amdgcn_tensor_load_to_lds(g0, g1, z4, z4, z8, 0);
#else
  __builtin_amdgcn_tensor_load_to_lds(g0, g1, z4, z4, 0);
#endif
}
#endif

// ---------------- manual staging (fallback + transposed B) ----------------
__device__ __forceinline__ void stage_rowmajor_f32(float* dst, const float* src,
                                                   int ld, int rows) {
  const int n4 = rows * 8;                     // float4 count (rows*32/4)
  for (int i = threadIdx.x; i < n4; i += 256) {
    int row = i >> 3;
    int col = (i & 7) << 2;
    *(float4*)(dst + row * 32 + col) = *(const float4*)(src + (long long)row * ld + col);
  }
}

// B is K x N row-major (32 k-rows x cols): store transposed into lds[n*32 + k].
__device__ __forceinline__ void stage_colmajor_f32(float* dst, const float* src,
                                                   int ld, int cols) {
  const int q = cols >> 2;
  for (int i = threadIdx.x; i < 32 * q; i += 256) {
    int k  = i / q;
    int n0 = (i - k * q) << 2;
    const float4 f = *(const float4*)(src + (long long)k * ld + n0);
    dst[(n0 + 0) * 32 + k] = f.x;
    dst[(n0 + 1) * 32 + k] = f.y;
    dst[(n0 + 2) * 32 + k] = f.z;
    dst[(n0 + 3) * 32 + k] = f.w;
  }
}

// f32 LDS -> f16 fragment (16-bit A-matrix 16x32 layout, ISA 7.12.2):
// lane l -> row l&15, K = (l>>4)*8 + {0..7} and +16.
__device__ __forceinline__ v16h frag_from_lds(const float* p) {
  float4 a0 = *(const float4*)(p);
  float4 a1 = *(const float4*)(p + 4);
  float4 b0 = *(const float4*)(p + 16);
  float4 b1 = *(const float4*)(p + 20);
  v16h f;
  f[0]  = (_Float16)a0.x; f[1]  = (_Float16)a0.y; f[2]  = (_Float16)a0.z; f[3]  = (_Float16)a0.w;
  f[4]  = (_Float16)a1.x; f[5]  = (_Float16)a1.y; f[6]  = (_Float16)a1.z; f[7]  = (_Float16)a1.w;
  f[8]  = (_Float16)b0.x; f[9]  = (_Float16)b0.y; f[10] = (_Float16)b0.z; f[11] = (_Float16)b0.w;
  f[12] = (_Float16)b1.x; f[13] = (_Float16)b1.y; f[14] = (_Float16)b1.z; f[15] = (_Float16)b1.w;
  return f;
}

// ---------------- TDM double-buffered, LDS-tiled WMMA GEMM ----------------
// C[m,n] = alpha * sum_k A[m,k] * B'[k,n]
// BCOL=false: B is N x K row-major (NT).  BCOL=true: B is K x N row-major (NN).
// MODE: 0=store 1=add(residual) 2=silu-store 3=multiply-into.
template<int MODE, bool BCOL, int WM, int WN, int TM, int TN>
__global__ __launch_bounds__(256)
void gemm_tile_k(const float* __restrict__ A, int lda, long long sAb, long long sAh,
                 const float* __restrict__ Bm, int ldb, long long sBb, long long sBh,
                 float* __restrict__ C, int ldc, long long sCb, long long sCh,
                 int K, float alpha, int ZH) {
  constexpr int BM = WM * TM * 16;
  constexpr int BN = WN * TN * 16;
  __shared__ float lA[2][BM * 32];
  __shared__ float lB[2][BN * 32];

  int z  = blockIdx.z;
  int zb = z / ZH, zh = z - zb * ZH;
  A  += (long long)zb * sAb + (long long)zh * sAh;
  Bm += (long long)zb * sBb + (long long)zh * sBh;
  C  += (long long)zb * sCb + (long long)zh * sCh;

  const float* Ag = A + (long long)blockIdx.y * BM * lda;
  const float* Bg = BCOL ? (Bm + blockIdx.x * BN)
                         : (Bm + (long long)blockIdx.x * BN * ldb);

  const int wave = threadIdx.x >> 5;
  const int wm = wave / WN, wn = wave - wm * WN;
  const int lane = threadIdx.x & 31;
  const int r  = lane & 15;
  const int kb = (lane >> 4) << 3;

  auto stage = [&](float* dA, float* dB, int k0) {
#if USE_TDM
    if (threadIdx.x == 0) {
      tdm_load_2d(dA, Ag + k0, BM, lda);                       // async DMA -> LDS
      if (!BCOL) tdm_load_2d(dB, Bg + k0, BN, ldb);
    }
    if (BCOL) stage_colmajor_f32(dB, Bg + (long long)k0 * ldb, ldb, BN);
#else
    stage_rowmajor_f32(dA, Ag + k0, lda, BM);
    if (BCOL) stage_colmajor_f32(dB, Bg + (long long)k0 * ldb, ldb, BN);
    else      stage_rowmajor_f32(dB, Bg + k0, ldb, BN);
#endif
  };
  auto wait_sync = [&]() {
#if USE_TDM
    if (threadIdx.x == 0) __builtin_amdgcn_s_wait_tensorcnt(0);
#endif
    __syncthreads();
  };

  v8f acc[TM][TN] = {};

  stage(lA[0], lB[0], 0);         // preload tile 0
  wait_sync();
  int cur = 0;

  for (int k0 = 0; k0 < K; k0 += 32) {
    int nxt = cur ^ 1;
    if (k0 + 32 < K) stage(lA[nxt], lB[nxt], k0 + 32);   // DMA next tile during compute

    v16h af[TM], bf[TN];
#pragma unroll
    for (int t = 0; t < TM; ++t)
      af[t] = frag_from_lds(&lA[cur][((wm * TM + t) * 16 + r) * 32 + kb]);
#pragma unroll
    for (int t = 0; t < TN; ++t)
      bf[t] = frag_from_lds(&lB[cur][((wn * TN + t) * 16 + r) * 32 + kb]);

#pragma unroll
    for (int i2 = 0; i2 < TM; ++i2)
#pragma unroll
      for (int j2 = 0; j2 < TN; ++j2)
        acc[i2][j2] = __builtin_amdgcn_wmma_f32_16x16x32_f16(
            false, af[i2], false, bf[j2], (short)0, acc[i2][j2], false, false);

    wait_sync();                  // next tile landed; all reads of cur complete
    cur = nxt;
  }

  // C/D layout: lane = N + 16*(M>=8), vgpr r = M mod 8 (ISA 7.12.2)
  const int n  = lane & 15;
  const int mb = (lane >> 4) << 3;
#pragma unroll
  for (int i2 = 0; i2 < TM; ++i2) {
    const long long mrow = (long long)blockIdx.y * BM + (wm * TM + i2) * 16 + mb;
#pragma unroll
    for (int j2 = 0; j2 < TN; ++j2) {
      const long long ncol = (long long)blockIdx.x * BN + (wn * TN + j2) * 16 + n;
      float* Cp = C + mrow * ldc + ncol;
#pragma unroll
      for (int rr = 0; rr < 8; ++rr) {
        float v = acc[i2][j2][rr] * alpha;
        float* dst = Cp + (long long)rr * ldc;
        if      (MODE == 0) *dst = v;
        else if (MODE == 1) *dst += v;
        else if (MODE == 2) *dst = v / (1.0f + expf(-v));
        else                *dst *= v;
      }
    }
  }
}

// ---------------- small kernels ----------------
__global__ void embed_k(const int* __restrict__ x, const float* __restrict__ emb,
                        float* __restrict__ h, long long n) {
  long long i = (long long)blockIdx.x * blockDim.x + threadIdx.x;
  if (i >= n) return;
  long long row = i / DDM;
  int d = (int)(i - row * DDM);
  h[i] = emb[(long long)x[row] * DDM + d];
}

__global__ void rmsnorm_k(const float* __restrict__ x, const float* __restrict__ w,
                          float* __restrict__ out) {
  __shared__ float red[256];
  long long row = blockIdx.x;
  const float* xr = x + row * DDM;
  float s = 0.f;
  for (int d = threadIdx.x; d < DDM; d += 256) { float v = xr[d]; s += v * v; }
  red[threadIdx.x] = s; __syncthreads();
  for (int st = 128; st > 0; st >>= 1) {
    if ((int)threadIdx.x < st) red[threadIdx.x] += red[threadIdx.x + st];
    __syncthreads();
  }
  float scale = rsqrtf(red[0] / (float)DDM + 1e-5f);
  for (int d = threadIdx.x; d < DDM; d += 256) out[row * DDM + d] = xr[d] * scale * w[d];
}

__global__ void rope_k(float* __restrict__ q, float* __restrict__ k) {
  int i = blockIdx.x * blockDim.x + threadIdx.x;
  const int total = BB * SSQ * NH * (HDIM / 2);
  if (i >= total) return;
  int d2 = i & 31; int t = i >> 5;
  int hh = t % NH; t /= NH;
  int s  = t % SSQ; int b = t / SSQ;
  float inv = expf(-(2.0f * d2) * (logf(10000.0f) / (float)HDIM));
  float fr = (float)s * inv;
  float cs = cosf(fr), sn = sinf(fr);
  long long base = (((long long)b * SSQ + s) * DDM) + hh * HDIM + 2 * d2;
  float qr = q[base], qi = q[base + 1];
  q[base] = qr * cs - qi * sn; q[base + 1] = qr * sn + qi * cs;
  float kr = k[base], ki = k[base + 1];
  k[base] = kr * cs - ki * sn; k[base + 1] = kr * sn + ki * cs;
}

__global__ void softmax_causal_k(float* __restrict__ sc) {
  __shared__ float red[256];
  long long row = blockIdx.x;                 // (b*H + h)*S + q
  int qi = (int)(row % SSQ);
  float* p = sc + row * SSQ;
  float mx = -3.0e38f;
  for (int kk = threadIdx.x; kk < SSQ; kk += 256)
    if (kk <= qi) mx = fmaxf(mx, p[kk]);
  red[threadIdx.x] = mx; __syncthreads();
  for (int st = 128; st > 0; st >>= 1) {
    if ((int)threadIdx.x < st) red[threadIdx.x] = fmaxf(red[threadIdx.x], red[threadIdx.x + st]);
    __syncthreads();
  }
  mx = red[0]; __syncthreads();
  float sum = 0.f;
  for (int kk = threadIdx.x; kk < SSQ; kk += 256) {
    float e = (kk <= qi) ? expf(p[kk] - mx) : 0.f;
    p[kk] = e; sum += e;
  }
  red[threadIdx.x] = sum; __syncthreads();
  for (int st = 128; st > 0; st >>= 1) {
    if ((int)threadIdx.x < st) red[threadIdx.x] += red[threadIdx.x + st];
    __syncthreads();
  }
  float invs = 1.0f / red[0];
  for (int kk = threadIdx.x; kk < SSQ; kk += 256) p[kk] *= invs;
}

// logits[b,t,v] += sum_{d=1..8} exp(-cw*(8-d)) * c[b,t-d,v]
__global__ void convadd_k(float* __restrict__ logits, const float* __restrict__ c,
                          const float* __restrict__ conv_w, long long n) {
  long long i = (long long)blockIdx.x * blockDim.x + threadIdx.x;
  if (i >= n) return;
  int v = (int)(i % VVC);
  long long t0 = i / VVC;
  int t = (int)(t0 % SSQ); int b = (int)(t0 / SSQ);
  float cw = conv_w[0];
  float acc = 0.f;
#pragma unroll
  for (int d = 1; d <= FCSN; ++d) {
    int tp = t - d;
    if (tp >= 0)
      acc += expf(-cw * (float)(FCSN - d)) * c[((long long)b * SSQ + tp) * (long long)VVC + v];
  }
  logits[i] += acc;
}

__global__ void zero_accum_k(float* a) { if (threadIdx.x < 4) a[threadIdx.x] = 0.f; }

__global__ void ce_loss_k(const float* __restrict__ logits, const int* __restrict__ y,
                          float* __restrict__ accum) {
  __shared__ float red[256];
  long long row = blockIdx.x;                 // b*S + t
  int b = (int)(row / SSQ), t = (int)(row % SSQ);
  const float* p = logits + row * (long long)VVC;
  float mx = -3.0e38f;
  for (int v = threadIdx.x; v < VVC; v += 256) mx = fmaxf(mx, p[v]);
  red[threadIdx.x] = mx; __syncthreads();
  for (int st = 128; st > 0; st >>= 1) {
    if ((int)threadIdx.x < st) red[threadIdx.x] = fmaxf(red[threadIdx.x], red[threadIdx.x + st]);
    __syncthreads();
  }
  mx = red[0]; __syncthreads();
  float sum = 0.f;
  for (int v = threadIdx.x; v < VVC; v += 256) sum += expf(p[v] - mx);
  red[threadIdx.x] = sum; __syncthreads();
  for (int st = 128; st > 0; st >>= 1) {
    if ((int)threadIdx.x < st) red[threadIdx.x] += red[threadIdx.x + st];
    __syncthreads();
  }
  if (threadIdx.x == 0) {
    int yt = y[(long long)b * YLD + t];
    if (yt >= 0 && yt < VVC) {
      float lse = mx + logf(red[0]);
      atomicAdd(&accum[0], lse - p[yt]);
      atomicAdd(&accum[1], 1.0f);
    }
  }
}

__device__ __forceinline__ float log_sigmoid_f(float z) {
  return fminf(z, 0.f) - log1pf(expf(-fabsf(z)));
}

__global__ void ctx_loss_k(const float* __restrict__ c, const int* __restrict__ y,
                           float* __restrict__ accum) {
  __shared__ float red[256];
  long long row = blockIdx.x;                 // b*S + s
  int b = (int)(row / SSQ), s = (int)(row % SSQ);
  int yc0[FCSN], yc1[FCSN];
#pragma unroll
  for (int j = 0; j < FCSN; ++j) {
    int idx = (s - (FCSN - 1) + j) % SSQ; if (idx < 0) idx += SSQ;
    yc0[j] = y[0 * YLD + (FCSN - 1) + idx];
    yc1[j] = y[1 * YLD + (FCSN - 1) + idx];
  }
  const int* ycb = (b == 0) ? yc0 : yc1;
  const float* p = c + row * (long long)VVC;
  float acc = 0.f;
  for (int v = threadIdx.x; v < VVC; v += 256) {
    float cv = p[v];
    int tgt = 0, c0 = 0, c1 = 0;
#pragma unroll
    for (int j = 0; j < FCSN; ++j) {
      tgt |= (ycb[j] == v);
      c0  += (yc0[j] == v);
      c1  += (yc1[j] == v);
    }
    float wgt = 1.0f + ((c0 > 1) || (c1 > 1) ? 0.5f : 0.0f);
    float bce = -(tgt ? log_sigmoid_f(cv) : log_sigmoid_f(-cv));
    acc += wgt * bce;
  }
  red[threadIdx.x] = acc; __syncthreads();
  for (int st = 128; st > 0; st >>= 1) {
    if ((int)threadIdx.x < st) red[threadIdx.x] += red[threadIdx.x + st];
    __syncthreads();
  }
  if (threadIdx.x == 0)
    atomicAdd(&accum[2], red[0] / ((float)BB * (float)SSQ * (float)VVC));
}

__global__ void finalize_k(const float* __restrict__ accum, float* __restrict__ out_scalars) {
  out_scalars[0] = accum[0] / fmaxf(accum[1], 1.0f);
  out_scalars[1] = accum[2];
}

// ---------------- host-side GEMM launcher ----------------
template<int MODE, bool BCOL, int WM, int WN, int TM, int TN>
static void launch_gemm(hipStream_t st,
                        const float* A, int lda, long long sAb, long long sAh,
                        const float* Bm, int ldb, long long sBb, long long sBh,
                        float* C, int ldc, long long sCb, long long sCh,
                        int M, int N, int K, float alpha, int nz, int ZH) {
  constexpr int BM = WM * TM * 16;
  constexpr int BN = WN * TN * 16;
  dim3 grd(N / BN, M / BM, nz);
  gemm_tile_k<MODE, BCOL, WM, WN, TM, TN><<<grd, dim3(256), 0, st>>>(
      A, lda, sAb, sAh, Bm, ldb, sBb, sBh, C, ldc, sCb, sCh, K, alpha, ZH);
}
// Big tiles: 128x128 (waves 2x4, 4x2 subtiles).  attn@V: 128x64 (waves 4x2, 2x2).
#define GEMM_BIG(MODE, ...) launch_gemm<MODE, false, 2, 4, 4, 2>(__VA_ARGS__)
#define GEMM_AV(...)        launch_gemm<0,    true,  4, 2, 2, 2>(__VA_ARGS__)

extern "C" void kernel_launch(void* const* d_in, const int* in_sizes, int n_in,
                              void* d_out, int out_size, void* d_ws, size_t ws_size,
                              hipStream_t stream) {
  (void)in_sizes; (void)n_in; (void)out_size; (void)ws_size;
  const int*   x       = (const int*)  d_in[0];
  const int*   y       = (const int*)  d_in[1];
  const float* emb     = (const float*)d_in[2];
  const float* wq      = (const float*)d_in[3];
  const float* wk      = (const float*)d_in[4];
  const float* wv      = (const float*)d_in[5];
  const float* wo      = (const float*)d_in[6];
  const float* w1      = (const float*)d_in[7];
  const float* w2      = (const float*)d_in[8];
  const float* w3      = (const float*)d_in[9];
  const float* anw     = (const float*)d_in[10];
  const float* fnw     = (const float*)d_in[11];
  const float* onw     = (const float*)d_in[12];
  const float* w_out   = (const float*)d_in[13];
  const float* w_ctx   = (const float*)d_in[14];
  const float* conv_w  = (const float*)d_in[15];

  float* logits = (float*)d_out;                      // [B,S,V] then loss, context_loss
  float* ws = (float*)d_ws;
  size_t off = 0;
  float* h   = ws + off; off += (size_t)MMR * DDM;
  float* a   = ws + off; off += (size_t)MMR * DDM;
  float* qb  = ws + off; off += (size_t)MMR * DDM;
  float* kb  = ws + off; off += (size_t)MMR * DDM;
  float* vb  = ws + off; off += (size_t)MMR * DDM;
  float* ob  = ws + off; off += (size_t)MMR * DDM;
  float* sc  = ws + off; off += (size_t)BB * NH * SSQ * SSQ;
  float* gu  = ws + off; off += (size_t)MMR * FFH;
  float* cb  = ws + off; off += (size_t)BB * SSQ * (size_t)VVC;
  float* accum = ws + off; off += 8;

  const long long SD  = (long long)SSQ * DDM;
  const long long HSS = (long long)NH * SSQ * SSQ;
  const long long SS2 = (long long)SSQ * SSQ;

  // embedding gather
  {
    long long n = (long long)MMR * DDM;
    embed_k<<<dim3((unsigned)((n + 255) / 256)), dim3(256), 0, stream>>>(x, emb, h, n);
  }
  zero_accum_k<<<dim3(1), dim3(32), 0, stream>>>(accum);

  for (int i = 0; i < LLN; ++i) {
    const float* wqi = wq + (size_t)i * DDM * DDM;
    const float* wki = wk + (size_t)i * DDM * DDM;
    const float* wvi = wv + (size_t)i * DDM * DDM;
    const float* woi = wo + (size_t)i * DDM * DDM;
    const float* w1i = w1 + (size_t)i * FFH * DDM;
    const float* w2i = w2 + (size_t)i * DDM * FFH;
    const float* w3i = w3 + (size_t)i * FFH * DDM;

    // a = rmsnorm(h) * attn_norm_w[i]
    rmsnorm_k<<<dim3(MMR), dim3(256), 0, stream>>>(h, anw + (size_t)i * DDM, a);

    // q/k/v = a @ w^T   (NT, f16 WMMA)
    GEMM_BIG(0, stream, a, DDM, 0,0, wqi, DDM, 0,0, qb, DDM, 0,0, MMR, DDM, DDM, 1.f, 1, 1);
    GEMM_BIG(0, stream, a, DDM, 0,0, wki, DDM, 0,0, kb, DDM, 0,0, MMR, DDM, DDM, 1.f, 1, 1);
    GEMM_BIG(0, stream, a, DDM, 0,0, wvi, DDM, 0,0, vb, DDM, 0,0, MMR, DDM, DDM, 1.f, 1, 1);

    // rope on q,k in place
    {
      int n = BB * SSQ * NH * (HDIM / 2);
      rope_k<<<dim3((n + 255) / 256), dim3(256), 0, stream>>>(qb, kb);
    }

    // scores[b,h] = q_head @ k_head^T / sqrt(hd)   (NT per (b,h), z = b*H + h)
    GEMM_BIG(0, stream, qb, DDM, SD, HDIM, kb, DDM, SD, HDIM,
             sc, SSQ, HSS, SS2, SSQ, SSQ, HDIM, 0.125f, BB * NH, NH);

    softmax_causal_k<<<dim3(BB * NH * SSQ), dim3(256), 0, stream>>>(sc);

    // o_head = attn @ v_head  (NN: B is K x N row-major with ld = D)
    GEMM_AV(stream, sc, SSQ, HSS, SS2, vb, DDM, SD, HDIM,
            ob, DDM, SD, HDIM, SSQ, HDIM, SSQ, 1.f, BB * NH, NH);

    // h += o @ wo^T   (residual fused)
    GEMM_BIG(1, stream, ob, DDM, 0,0, woi, DDM, 0,0, h, DDM, 0,0, MMR, DDM, DDM, 1.f, 1, 1);

    // FFN: f = rmsnorm(h); gu = silu(f@w1^T) * (f@w3^T); h += gu @ w2^T
    rmsnorm_k<<<dim3(MMR), dim3(256), 0, stream>>>(h, fnw + (size_t)i * DDM, a);
    GEMM_BIG(2, stream, a, DDM, 0,0, w1i, DDM, 0,0, gu, FFH, 0,0, MMR, FFH, DDM, 1.f, 1, 1);
    GEMM_BIG(3, stream, a, DDM, 0,0, w3i, DDM, 0,0, gu, FFH, 0,0, MMR, FFH, DDM, 1.f, 1, 1);
    GEMM_BIG(1, stream, gu, FFH, 0,0, w2i, FFH, 0,0, h, DDM, 0,0, MMR, DDM, FFH, 1.f, 1, 1);
  }

  // final norm + heads
  rmsnorm_k<<<dim3(MMR), dim3(256), 0, stream>>>(h, onw, a);
  GEMM_BIG(0, stream, a, DDM, 0,0, w_out, DDM, 0,0, logits, VVC, 0,0, MMR, VVC, DDM, 1.f, 1, 1);
  GEMM_BIG(0, stream, a, DDM, 0,0, w_ctx, DDM, 0,0, cb,     VVC, 0,0, MMR, VVC, DDM, 1.f, 1, 1);

  // logits += causal exponential conv of context head
  {
    long long n = (long long)BB * SSQ * VVC;
    convadd_k<<<dim3((unsigned)((n + 255) / 256)), dim3(256), 0, stream>>>(logits, cb, conv_w, n);
  }

  // losses
  ce_loss_k<<<dim3(MMR), dim3(256), 0, stream>>>(logits, y, accum);
  ctx_loss_k<<<dim3(MMR), dim3(256), 0, stream>>>(cb, y, accum);
  finalize_k<<<dim3(1), dim3(1), 0, stream>>>(accum, logits + (size_t)MMR * VVC);
}
